// Model_52518860096358
// MI455X (gfx1250) — matrix operand order repaired and checked by reading, not compiled, not run on hardware
//
#include <hip/hip_runtime.h>

typedef __attribute__((ext_vector_type(16))) __bf16 v16bf;
typedef __attribute__((ext_vector_type(8)))  __bf16 v8bf;
typedef __attribute__((ext_vector_type(8)))  float  v8f;

#define B_   8
#define SQ_  4096
#define SK_  4096
#define D_   64
#define BM   128          // queries per workgroup (8 waves x 16 rows)
#define BN   128          // keys per iteration
#define LDK  72           // K-tile row stride (d-major), bf16 elems: 144B rows
#define LDV  136          // V^T row stride (key-major), bf16 elems: 272B rows
#define LDP  136          // P row stride (key-major)

// ---- float -> bf16, round to nearest even ----
__device__ __forceinline__ unsigned short f2bf_u(float f) {
  unsigned int u = __builtin_bit_cast(unsigned int, f);
  u += 0x7FFFu + ((u >> 16) & 1u);
  return (unsigned short)(u >> 16);
}
__device__ __forceinline__ __bf16 f2bf(float f) {
  unsigned short s = f2bf_u(f);
  return __builtin_bit_cast(__bf16, s);
}
__device__ __forceinline__ unsigned int pack2(float a, float b) {
  return (unsigned int)f2bf_u(a) | ((unsigned int)f2bf_u(b) << 16);
}

// 16 contiguous bf16 (B-operand fragment slice) — two 16B LDS loads
__device__ __forceinline__ v16bf load16(const __bf16* p) {
  v8bf lo = *(const v8bf*)p;
  v8bf hi = *(const v8bf*)(p + 8);
  return __builtin_shufflevector(lo, hi, 0,1,2,3,4,5,6,7,8,9,10,11,12,13,14,15);
}
// A-operand fragment slice: elements [base..base+7] and [base+16..base+23]
__device__ __forceinline__ v16bf loadA(const __bf16* p) {
  v8bf lo = *(const v8bf*)p;
  v8bf hi = *(const v8bf*)(p + 16);
  return __builtin_shufflevector(lo, hi, 0,1,2,3,4,5,6,7,8,9,10,11,12,13,14,15);
}

__global__ __launch_bounds__(256, 2)
void Model_52518860096358_fattn(const float* __restrict__ Q,
                                const float* __restrict__ K,
                                const float* __restrict__ V,
                                const float* __restrict__ scaleDiv,
                                float* __restrict__ Out) {
  __shared__ __bf16 Ks[BN * LDK];        // K tile, row-major [key][d]
  __shared__ __bf16 Vt[D_ * LDV];        // V tile, transposed [d][key]
  __shared__ __bf16 Pw[8 * 16 * LDP];    // per-wave P scratch [wave][row][key]

  const int tid  = threadIdx.x;
  const int wave = tid >> 5;
  const int lane = tid & 31;
  const int l15  = lane & 15;
  const int hig  = lane >> 4;            // 0 | 1 (half-wave)
  const int klo  = hig * 8;              // A-fragment K offset
  const int khi  = hig * 16;             // B-fragment K offset

  const int b  = blockIdx.x / (SQ_ / BM);
  const int mt = blockIdx.x % (SQ_ / BM);
  const int q0 = mt * BM + wave * 16;    // first query row owned by this wave

  const float c = 1.44269504088896340736f / scaleDiv[0];  // log2(e)/scale

  // ---- Q A-fragments, pre-scaled by log2(e)/scale, bf16, resident ----
  v16bf aq[2];
  {
    const float* qb = Q + ((size_t)b * SQ_ + (size_t)(q0 + l15)) * D_;
#pragma unroll
    for (int kk = 0; kk < 2; ++kk) {
      const float* qp = qb + kk * 32 + klo;
      v16bf a;
#pragma unroll
      for (int j = 0; j < 8; ++j) {
        a[j]     = f2bf(qp[j] * c);
        a[8 + j] = f2bf(qp[16 + j] * c);
      }
      aq[kk] = a;
    }
  }

  v8f o[4];
#pragma unroll
  for (int nt = 0; nt < 4; ++nt) o[nt] = {};
  float mrow[8], lrow[8];
#pragma unroll
  for (int r = 0; r < 8; ++r) { mrow[r] = -1e30f; lrow[r] = 0.0f; }

  __bf16* Pmine = Pw + wave * 16 * LDP;

  // staging decomposition
  const int krow = tid >> 1;             // K: 0..127
  const int kcol = (tid & 1) * 32;       // K: 0 or 32
  const int vkey = (tid >> 2) * 2;       // V: even key 0..126
  const int vcol = (tid & 3) * 16;       // V: 0,16,32,48

  for (int kt = 0; kt < SK_ / BN; ++kt) {
    const int kv0 = kt * BN;
    __syncthreads();                     // previous-iteration LDS reads done

    // ---- stage K (row-major bf16) ----
    {
      const float* kp = K + ((size_t)b * SK_ + (size_t)(kv0 + krow)) * D_ + kcol;
      float kf[32];
#pragma unroll
      for (int j = 0; j < 32; ++j) kf[j] = kp[j];
      unsigned int* kd = (unsigned int*)&Ks[krow * LDK + kcol];
#pragma unroll
      for (int j = 0; j < 16; ++j) kd[j] = pack2(kf[2*j], kf[2*j+1]);
      if (kt + 1 < SK_ / BN) __builtin_prefetch(kp + (size_t)BN * D_, 0, 1);
    }
    // ---- stage V transposed (key-pairs packed as b32) ----
    {
      const float* vp0 = V + ((size_t)b * SK_ + (size_t)(kv0 + vkey)) * D_ + vcol;
      const float* vp1 = vp0 + D_;
      float v0[16], v1[16];
#pragma unroll
      for (int j = 0; j < 16; ++j) { v0[j] = vp0[j]; v1[j] = vp1[j]; }
#pragma unroll
      for (int j = 0; j < 16; ++j)
        *(unsigned int*)&Vt[(vcol + j) * LDV + vkey] = pack2(v0[j], v1[j]);
      if (kt + 1 < SK_ / BN) __builtin_prefetch(vp0 + (size_t)BN * D_, 0, 1);
    }
    __syncthreads();

    // ---- S = (Q*c) @ K^T : 16x v_wmma_f32_16x16x32_bf16 ----
    v8f s[8];
#pragma unroll
    for (int nt = 0; nt < 8; ++nt) {
      v8f acc = {};
#pragma unroll
      for (int kk = 0; kk < 2; ++kk) {
        v16bf bk = loadA(&Ks[(nt * 16 + l15) * LDK + kk * 32 + klo]);
        acc = __builtin_amdgcn_wmma_f32_16x16x32_bf16(false, aq[kk], false, bk,
                                                      (short)0, acc, false, false);
      }
      s[nt] = acc;
    }

    // ---- online softmax (scale already folded into Q; exp2 domain) ----
#pragma unroll
    for (int r = 0; r < 8; ++r) {
      float m = s[0][r];
#pragma unroll
      for (int nt = 1; nt < 8; ++nt) m = fmaxf(m, s[nt][r]);
#pragma unroll
      for (int d = 1; d < 16; d <<= 1) m = fmaxf(m, __shfl_xor(m, d, 32));
      const float mn  = fmaxf(mrow[r], m);
      const float fac = exp2f(mrow[r] - mn);
      mrow[r] = mn;
      float rs = 0.0f;
#pragma unroll
      for (int nt = 0; nt < 8; ++nt) {
        const float p = exp2f(s[nt][r] - mn);
        s[nt][r] = p;
        rs += p;
      }
#pragma unroll
      for (int d = 1; d < 16; d <<= 1) rs += __shfl_xor(rs, d, 32);
      lrow[r] = lrow[r] * fac + rs;
#pragma unroll
      for (int nt = 0; nt < 4; ++nt) o[nt][r] *= fac;
    }

    // ---- P: C-layout -> bf16 LDS -> A-layout ----
#pragma unroll
    for (int r = 0; r < 8; ++r) {
      const int prow = r + 8 * hig;
#pragma unroll
      for (int nt = 0; nt < 8; ++nt)
        Pmine[prow * LDP + nt * 16 + l15] = f2bf(s[nt][r]);
    }
    __syncthreads();

    v16bf ap[4];
#pragma unroll
    for (int kk = 0; kk < 4; ++kk)
      ap[kk] = loadA(&Pmine[l15 * LDP + kk * 32 + klo]);

    // ---- O += P @ V : 16x v_wmma_f32_16x16x32_bf16 ----
#pragma unroll
    for (int nt = 0; nt < 4; ++nt) {
#pragma unroll
      for (int kk = 0; kk < 4; ++kk) {
        v16bf bv = loadA(&Vt[(nt * 16 + l15) * LDV + kk * 32 + klo]);
        o[nt] = __builtin_amdgcn_wmma_f32_16x16x32_bf16(false, ap[kk], false, bv,
                                                        (short)0, o[nt], false, false);
      }
    }
  }

  // ---- epilogue: normalize and store fp32 ----
#pragma unroll
  for (int r = 0; r < 8; ++r) {
    const float inv = 1.0f / lrow[r];
    const int q = q0 + r + 8 * hig;
    float* op = Out + ((size_t)b * SQ_ + (size_t)q) * D_;
#pragma unroll
    for (int nt = 0; nt < 4; ++nt)
      op[nt * 16 + l15] = o[nt][r] * inv;
  }
}

extern "C" void kernel_launch(void* const* d_in, const int* in_sizes, int n_in,
                              void* d_out, int out_size, void* d_ws, size_t ws_size,
                              hipStream_t stream) {
  (void)in_sizes; (void)n_in; (void)out_size; (void)d_ws; (void)ws_size;
  const float* Q  = (const float*)d_in[0];
  const float* K  = (const float*)d_in[1];
  const float* V  = (const float*)d_in[2];
  // d_in[3] = dropout p (static 0) — unused
  const float* sc = (const float*)d_in[4];
  float* Out = (float*)d_out;

  dim3 grid(B_ * (SQ_ / BM));
  dim3 block(256);
  Model_52518860096358_fattn<<<grid, block, 0, stream>>>(Q, K, V, sc, Out);
}
